// GCN_1795296329933
// MI455X (gfx1250) — matrix operand order repaired and checked by reading, not compiled, not run on hardware
//
#include <hip/hip_runtime.h>
#include <hip/hip_bf16.h>
#include <math.h>

typedef __attribute__((ext_vector_type(16))) _Float16 v16h;
typedef __attribute__((ext_vector_type(4)))  _Float16 v4h;
typedef __attribute__((ext_vector_type(8)))  float    v8f;
typedef __attribute__((ext_vector_type(4)))  float    v4f;

#define NN 50000
#define NE 1600000

__device__ __forceinline__ void atomAddF(float* p, float v) {
    __hip_atomic_fetch_add(p, v, __ATOMIC_RELAXED, __HIP_MEMORY_SCOPE_AGENT);
}

// ---------------- normalization precompute ----------------

__global__ void k_init_deg(float* deg, int n) {
    int i = blockIdx.x * blockDim.x + threadIdx.x;
    if (i < n) deg[i] = 1.0f;                       // self-loop weight 1
}

__global__ void k_accum_deg(const int* __restrict__ dst, const float* __restrict__ w,
                            float* deg, int e) {
    int i = blockIdx.x * blockDim.x + threadIdx.x;
    if (i < e) atomAddF(&deg[dst[i]], w[i]);
}

__global__ void k_dinv(float* deg, float* selfnorm, int n) {
    int i = blockIdx.x * blockDim.x + threadIdx.x;
    if (i < n) {
        float d = deg[i];
        float r = (d > 0.0f) ? rsqrtf(fmaxf(d, 1e-30f)) : 0.0f;
        deg[i] = r;               // deg buffer becomes dinv
        selfnorm[i] = r * r;      // self-loop norm: dinv*1*dinv
    }
}

__global__ void k_edge_norm(const int* __restrict__ src, const int* __restrict__ dst,
                            const float* __restrict__ w, const float* __restrict__ dinv,
                            float* __restrict__ norm, int e) {
    int i = blockIdx.x * blockDim.x + threadIdx.x;
    if (i < e) norm[i] = dinv[src[i]] * w[i] * dinv[dst[i]];
}

// ---------------- GEMM (WMMA f32 <- f16 x f16) ----------------

// Wt[n*128 + k] = (f16) W[k*NO + n]   (column-major f16 copy of W, K=128 fixed)
__global__ void k_convW(const float* __restrict__ W, _Float16* __restrict__ Wt, int NO) {
    int i = blockIdx.x * blockDim.x + threadIdx.x;
    if (i < 128 * NO) {
        int k = i % 128, n = i / 128;
        Wt[(size_t)n * 128 + k] = (_Float16)W[(size_t)k * NO + n];
    }
}

// fused activation + fp32 -> f16 convert (GEMM A operand producer)
// kind 0: elu, 1: sigmoid, 2: identity
__global__ void k_cvt_act(const float* __restrict__ in, _Float16* __restrict__ out16,
                          int n4, int kind) {
    int i = blockIdx.x * blockDim.x + threadIdx.x;   // one float4 per thread
    if (i >= n4) return;
    v4f v = *(const v4f*)(in + (size_t)i * 4);
    v4h o;
#pragma unroll
    for (int j = 0; j < 4; ++j) {
        float t = v[j];
        if (kind == 0)      t = (t > 0.0f) ? t : expm1f(t);
        else if (kind == 1) t = 1.0f / (1.0f + expf(-t));
        o[j] = (_Float16)t;
    }
    *(v4h*)(out16 + (size_t)i * 4) = o;
}

// C[M x NOUT] = A16[M x 128] * W, via v_wmma_f32_16x16x32_f16. M % 16 == 0.
template<int NOUT>
__global__ __launch_bounds__(256) void k_gemm(const _Float16* __restrict__ A16,
                                              const _Float16* __restrict__ Bt,
                                              float* __restrict__ C, int M) {
    constexpr int TN = NOUT / 16;
    const int wave = threadIdx.x >> 5;
    const int lane = threadIdx.x & 31;
    const int tileLin = blockIdx.x * 8 + wave;
    const int tilesM = M >> 4;
    const int tm = tileLin / TN;
    const int tn = tileLin % TN;
    if (tm >= tilesM) return;                       // wave-uniform guard (EXEC all-1 inside)

    const int mrow = (lane & 15);
    const int ksel = lane >> 4;                     // 0: lanes 0-15, 1: lanes 16-31
    const _Float16* ap = A16 + (size_t)(tm * 16 + mrow) * 128 + ksel * 16;
    const _Float16* bp = Bt  + (size_t)(tn * 16 + mrow) * 128 + ksel * 16;

    v8f c = {};
#pragma unroll
    for (int k0 = 0; k0 < 128; k0 += 32) {
        v16h a, b;
#pragma unroll
        for (int i = 0; i < 8; ++i) {               // contiguous K layout (matches B)
            a[i]     = ap[k0 + i];                  // K = k0 + ksel*16 + i
            a[i + 8] = ap[k0 + 8 + i];              // K = k0 + ksel*16 + 8 + i
        }
        b = *(const v16h*)(bp + k0);                // K = k0 + ksel*16 .. +15, col = tn*16+mrow
        c = __builtin_amdgcn_wmma_f32_16x16x32_f16(false, a, false, b, (short)0, c,
                                                   false, false);
    }
    // D layout: VGPR r -> row = r + 8*ksel, col = lane&15
    float* cp = C + (size_t)(tm * 16 + ksel * 8) * NOUT + tn * 16 + mrow;
#pragma unroll
    for (int r = 0; r < 8; ++r) cp[(size_t)r * NOUT] = c[r];
}

// ---------------- aggregation ----------------

// out[n][f] = selfnorm[n] * h[n][f] + bias[f]
template<int NOUT>
__global__ void k_init_out(const float* __restrict__ h, const float* __restrict__ selfnorm,
                           const float* __restrict__ bias, float* __restrict__ out, int n) {
    constexpr int T = NOUT / 4;
    int tid = blockIdx.x * blockDim.x + threadIdx.x;
    int node = tid / T, f = (tid % T) * 4;
    if (node >= n) return;
    float s = selfnorm[node];
    v4f hv = *(const v4f*)(h + (size_t)node * NOUT + f);
    v4f o;
#pragma unroll
    for (int i = 0; i < 4; ++i) o[i] = s * hv[i] + bias[f + i];
    *(v4f*)(out + (size_t)node * NOUT + f) = o;
}

// out[dst[e]][f] += norm[e] * h[src[e]][f]
template<int NOUT>
__global__ void k_scatter(const float* __restrict__ h, const int* __restrict__ src,
                          const int* __restrict__ dst, const float* __restrict__ norm,
                          float* __restrict__ out, int e) {
    constexpr int T = NOUT / 4;
    int tid = blockIdx.x * blockDim.x + threadIdx.x;
    int edge = tid / T, f = (tid % T) * 4;
    if (edge >= e) return;
    int s = src[edge], d = dst[edge];
    float w = norm[edge];
    v4f hv = *(const v4f*)(h + (size_t)s * NOUT + f);
    float* op = out + (size_t)d * NOUT + f;
    atomAddF(op + 0, hv[0] * w);
    atomAddF(op + 1, hv[1] * w);
    atomAddF(op + 2, hv[2] * w);
    atomAddF(op + 3, hv[3] * w);
}

__global__ void k_logsoftmax16(float* out, int n) {
    int i = blockIdx.x * blockDim.x + threadIdx.x;
    if (i >= n) return;
    float* p = out + (size_t)i * 16;
    float m = p[0];
#pragma unroll
    for (int j = 1; j < 16; ++j) m = fmaxf(m, p[j]);
    float s = 0.0f;
#pragma unroll
    for (int j = 0; j < 16; ++j) s += expf(p[j] - m);
    float ls = m + logf(s);
#pragma unroll
    for (int j = 0; j < 16; ++j) p[j] = p[j] - ls;
}

// ---------------- host orchestration ----------------

static inline int cdiv(long long a, int b) { return (int)((a + b - 1) / b); }

extern "C" void kernel_launch(void* const* d_in, const int* in_sizes, int n_in,
                              void* d_out, int out_size, void* d_ws, size_t ws_size,
                              hipStream_t stream) {
    const float* x   = (const float*)d_in[0];
    const int*   ei  = (const int*)  d_in[1];
    const float* ew  = (const float*)d_in[2];
    const float* W1  = (const float*)d_in[3];  const float* b1 = (const float*)d_in[4];
    const float* W2  = (const float*)d_in[5];  const float* b2 = (const float*)d_in[6];
    const float* W3  = (const float*)d_in[7];  const float* b3 = (const float*)d_in[8];
    const float* W4  = (const float*)d_in[9];  const float* b4 = (const float*)d_in[10];
    float* out = (float*)d_out;

    const int* src = ei;            // edge_index[0]
    const int* dst = ei + NE;       // edge_index[1]

    // carve workspace
    char* ws = (char*)d_ws;
    size_t off = 0;
    auto carve = [&](size_t bytes) -> void* {
        void* p = ws + off;
        off += (bytes + 255) & ~(size_t)255;
        return p;
    };
    float*    dinv     = (float*)   carve((size_t)NN * 4);
    float*    selfnorm = (float*)   carve((size_t)NN * 4);
    float*    norm     = (float*)   carve((size_t)NE * 4);
    float*    bufA     = (float*)   carve((size_t)NN * 128 * 4);   // GEMM result h (fp32)
    float*    bufB     = (float*)   carve((size_t)NN * 128 * 4);   // aggregated layer output
    _Float16* hf16     = (_Float16*)carve((size_t)NN * 128 * 2);   // activated input, f16 (A)
    _Float16* Wt       = (_Float16*)carve((size_t)128 * 128 * 2);  // f16 transposed weights

    // ---- gcn_norm ----
    k_init_deg <<<cdiv(NN, 256), 256, 0, stream>>>(dinv, NN);
    k_accum_deg<<<cdiv(NE, 256), 256, 0, stream>>>(dst, ew, dinv, NE);
    k_dinv     <<<cdiv(NN, 256), 256, 0, stream>>>(dinv, selfnorm, NN);
    k_edge_norm<<<cdiv(NE, 256), 256, 0, stream>>>(src, dst, ew, dinv, norm, NE);

    const int gemm128_blocks = NN / 16;                            // 3125 (8 tiles/block)
    const int gemm16_blocks  = cdiv(NN / 16, 8);                   // 391
    const int cvt_blocks     = cdiv((long long)NN * 128 / 4, 256); // fp32->f16, float4/thread

    // ---- layer 1: x -> bufB  (GCNConv + elu deferred to next cvt) ----
    k_convW<<<cdiv(128 * 128, 256), 256, 0, stream>>>(W1, Wt, 128);
    k_cvt_act<<<cvt_blocks, 256, 0, stream>>>(x, hf16, NN * 128 / 4, 2);       // identity
    k_gemm<128><<<gemm128_blocks, 256, 0, stream>>>(hf16, Wt, bufA, NN);
    k_init_out<128><<<cdiv((long long)NN * 32, 256), 256, 0, stream>>>(bufA, selfnorm, b1, bufB, NN);
    k_scatter<128><<<cdiv((long long)NE * 32, 256), 256, 0, stream>>>(bufA, src, dst, norm, bufB, NE);

    // ---- layer 2: elu(bufB) -> bufB ----
    k_convW<<<cdiv(128 * 128, 256), 256, 0, stream>>>(W2, Wt, 128);
    k_cvt_act<<<cvt_blocks, 256, 0, stream>>>(bufB, hf16, NN * 128 / 4, 0);    // elu
    k_gemm<128><<<gemm128_blocks, 256, 0, stream>>>(hf16, Wt, bufA, NN);
    k_init_out<128><<<cdiv((long long)NN * 32, 256), 256, 0, stream>>>(bufA, selfnorm, b2, bufB, NN);
    k_scatter<128><<<cdiv((long long)NE * 32, 256), 256, 0, stream>>>(bufA, src, dst, norm, bufB, NE);

    // ---- layer 3: sigmoid(bufB) -> bufB ----
    k_convW<<<cdiv(128 * 128, 256), 256, 0, stream>>>(W3, Wt, 128);
    k_cvt_act<<<cvt_blocks, 256, 0, stream>>>(bufB, hf16, NN * 128 / 4, 1);    // sigmoid
    k_gemm<128><<<gemm128_blocks, 256, 0, stream>>>(hf16, Wt, bufA, NN);
    k_init_out<128><<<cdiv((long long)NN * 32, 256), 256, 0, stream>>>(bufA, selfnorm, b3, bufB, NN);
    k_scatter<128><<<cdiv((long long)NE * 32, 256), 256, 0, stream>>>(bufA, src, dst, norm, bufB, NE);

    // ---- layer 4: elu(bufB) -> d_out (16 classes), log_softmax ----
    k_convW<<<cdiv(128 * 16, 256), 256, 0, stream>>>(W4, Wt, 16);
    k_cvt_act<<<cvt_blocks, 256, 0, stream>>>(bufB, hf16, NN * 128 / 4, 0);    // elu
    k_gemm<16><<<gemm16_blocks, 256, 0, stream>>>(hf16, Wt, bufA, NN);
    k_init_out<16><<<cdiv((long long)NN * 4, 256), 256, 0, stream>>>(bufA, selfnorm, b4, out, NN);
    k_scatter<16><<<cdiv((long long)NE * 4, 256), 256, 0, stream>>>(bufA, src, dst, norm, out, NE);
    k_logsoftmax16<<<cdiv(NN, 256), 256, 0, stream>>>(out, NN);
}
